// PointNetSAModule_24764781429154
// MI455X (gfx1250) — matrix operand order repaired and checked by reading, not compiled, not run on hardware
//
#include <hip/hip_runtime.h>
#include <hip/hip_bf16.h>

// ---------------------------------------------------------------------------
// PointNet SA module for MI455X (gfx1250, wave32, WMMA).
// Pipeline: featT(bf16) -> FPS -> ball-query+gather/pack(bf16) ->
//           [WMMA bf16 GEMM + BN-stats] x3 with BN+ReLU between layers ->
//           max over K + transpose -> out.
// ---------------------------------------------------------------------------

#define Bn    8
#define Nn    8192
#define CIN   64
#define Mn    1024
#define Kn    64
#define RAD2  (0.2f * 0.2f)
#define Pn    (Bn * Mn * Kn)        // 524288 rows through the MLP

typedef __attribute__((ext_vector_type(16))) __bf16          v16bf;
typedef __attribute__((ext_vector_type(16))) unsigned short  v16u;
typedef __attribute__((ext_vector_type(8)))  unsigned short  v8u;
typedef __attribute__((ext_vector_type(8)))  float           v8f;

__device__ __forceinline__ unsigned short f2bf(float f) {
    unsigned u = __builtin_bit_cast(unsigned, f);
    unsigned r = u + 0x7FFFu + ((u >> 16) & 1u);   // round-to-nearest-even
    return (unsigned short)(r >> 16);
}
__device__ __forceinline__ float bf2f(unsigned short h) {
    unsigned u = ((unsigned)h) << 16;
    return __builtin_bit_cast(float, u);
}

// ---------------------------------------------------------------------------
// feat (B,C,N) fp32 -> featT (B,N,C) bf16   (coalesced gather source)
// ---------------------------------------------------------------------------
__global__ void feat_transpose_bf16(const float* __restrict__ feat,
                                    unsigned short* __restrict__ featT) {
    size_t i = (size_t)blockIdx.x * 256 + threadIdx.x;   // < B*C*N
    size_t n = i & (Nn - 1);
    size_t c = (i >> 13) & (CIN - 1);
    size_t b = i >> 19;
    featT[(b * Nn + n) * CIN + c] = f2bf(feat[i]);
}

// Pad + convert weights: wp[o][k] = bf16(w[o][k]) for k<Cin, else 0.
__global__ void pack_weights(const float* __restrict__ w,
                             unsigned short* __restrict__ wp,
                             int Cout, int Cin, int Cinp) {
    int i = blockIdx.x * 256 + threadIdx.x;
    if (i >= Cout * Cinp) return;
    int o = i / Cinp, k = i - o * Cinp;
    wp[i] = (k < Cin) ? f2bf(w[o * Cin + k]) : (unsigned short)0;
}

__global__ void copy_f32(const float* __restrict__ s, float* __restrict__ d, int n) {
    int i = blockIdx.x * 256 + threadIdx.x;
    if (i < n) d[i] = s[i];
}

__global__ void zero_f32(float* __restrict__ p, int n) {
    int i = blockIdx.x * 256 + threadIdx.x;
    if (i < n) p[i] = 0.0f;
}

// ---------------------------------------------------------------------------
// Furthest point sampling. One block per batch; dist[] lives in LDS.
// Matches jax scan: emits current 'far' (starting at 0) then updates.
// Argmax tie-break = lowest index (jnp.argmax).
// ---------------------------------------------------------------------------
__global__ __launch_bounds__(1024) void fps_kernel(const float* __restrict__ coords,
                                                   float* __restrict__ centersWS,  // (B,M,3)
                                                   float* __restrict__ outCenters) // (B,3,M)
{
    __shared__ float dist[Nn];
    __shared__ float red[1024];
    __shared__ int   redi[1024];
    __shared__ float cc[3];
    __shared__ int   cur;

    const int b = blockIdx.x, tid = threadIdx.x;
    const float* px = coords + (size_t)b * 3 * Nn;
    const float* py = px + Nn;
    const float* pz = py + Nn;

    for (int n = tid; n < Nn; n += 1024) dist[n] = 1e10f;
    if (tid == 0) cur = 0;
    __syncthreads();

    for (int m = 0; m < Mn; ++m) {
        if (tid == 0) {
            int c0 = cur;
            float x = px[c0], y = py[c0], z = pz[c0];
            cc[0] = x; cc[1] = y; cc[2] = z;
            size_t cmi = (size_t)b * Mn + m;
            centersWS[cmi * 3 + 0] = x;
            centersWS[cmi * 3 + 1] = y;
            centersWS[cmi * 3 + 2] = z;
            outCenters[((size_t)b * 3 + 0) * Mn + m] = x;
            outCenters[((size_t)b * 3 + 1) * Mn + m] = y;
            outCenters[((size_t)b * 3 + 2) * Mn + m] = z;
        }
        __syncthreads();
        float cx = cc[0], cy = cc[1], cz = cc[2];
        float best = -1.0f;
        int   bi = tid;
        for (int n = tid; n < Nn; n += 1024) {
            float dx = px[n] - cx, dy = py[n] - cy, dz = pz[n] - cz;
            float d = dx * dx + dy * dy + dz * dz;
            float nd = dist[n];
            nd = d < nd ? d : nd;
            dist[n] = nd;
            if (nd > best) { best = nd; bi = n; }   // '>' keeps lowest index on tie
        }
        red[tid] = best; redi[tid] = bi;
        __syncthreads();
        for (int s = 512; s > 0; s >>= 1) {
            if (tid < s) {
                float ov = red[tid + s]; int oi = redi[tid + s];
                if (ov > red[tid] || (ov == red[tid] && oi < redi[tid])) {
                    red[tid] = ov; redi[tid] = oi;
                }
            }
            __syncthreads();
        }
        if (tid == 0) cur = redi[0];
        __syncthreads();
    }
}

// ---------------------------------------------------------------------------
// Ball query + gather + pack X0 (bf16, 96-col rows: [rel(3) | feat(64) | 0(29)])
// One wave32 per center: ballot + prefix popcount keeps the first K in-radius
// points in original order; pads with the first hit (index 0 if none).
// ---------------------------------------------------------------------------
__global__ __launch_bounds__(256) void ball_pack(const float* __restrict__ coords,
                                                 const float* __restrict__ centersWS,
                                                 const unsigned short* __restrict__ featT,
                                                 unsigned short* __restrict__ X0) {
    __shared__ int selbuf[8][Kn];
    const int wave = threadIdx.x >> 5, lane = threadIdx.x & 31;
    const int cm = blockIdx.x * 8 + wave;
    const int b = cm >> 10;

    const float* px = coords + (size_t)b * 3 * Nn;
    const float* py = px + Nn;
    const float* pz = py + Nn;

    const float cx = centersWS[(size_t)cm * 3 + 0];
    const float cy = centersWS[(size_t)cm * 3 + 1];
    const float cz = centersWS[(size_t)cm * 3 + 2];

    int cnt = 0;
    for (int n0 = 0; n0 < Nn && cnt < Kn; n0 += 32) {
        int n = n0 + lane;
        float dx = px[n] - cx, dy = py[n] - cy, dz = pz[n] - cz;
        bool within = (dx * dx + dy * dy + dz * dz) < RAD2;
        unsigned mask = (unsigned)__ballot(within);
        int pos = cnt + __popc(mask & ((1u << lane) - 1u));
        if (within && pos < Kn) selbuf[wave][pos] = n;
        cnt += __popc(mask);
    }
    if (cnt > Kn) cnt = Kn;
    __threadfence_block();
    int firstIdx = (cnt > 0) ? selbuf[wave][0] : 0;
    for (int s = cnt + lane; s < Kn; s += 32) selbuf[wave][s] = firstIdx;
    __threadfence_block();

    const size_t rowBase = (size_t)cm * Kn;
    for (int k = 0; k < Kn; ++k) {
        int idx = selbuf[wave][k];
        unsigned short* xr = X0 + (rowBase + k) * 96;
        const unsigned short* fr = featT + ((size_t)b * Nn + idx) * CIN;
        if (lane < 3) {
            float d = (lane == 0) ? (px[idx] - cx)
                    : (lane == 1) ? (py[idx] - cy)
                                  : (pz[idx] - cz);
            xr[lane] = f2bf(d);
        } else {
            xr[lane] = fr[lane - 3];          // channels 0..28
        }
        xr[lane + 32] = fr[lane + 29];        // channels 29..60
        xr[lane + 64] = (lane < 3) ? fr[lane + 61] : (unsigned short)0;  // 61..63, pad
    }
}

// ---------------------------------------------------------------------------
// WMMA bf16 GEMM  Y[P x COUT] = X[P x CINP] * W^T  (+bias), fused BN stats.
// Block = 8 waves; wave computes 16 rows x 64 cols (4 accum tiles).
// A/B fragments both use contiguous k = 16*grp + i order (16-bit 16x32 A, 32x16 B).
// ---------------------------------------------------------------------------
template <int CINP, int COUT>
__global__ __launch_bounds__(256) void gemm_bn(const unsigned short* __restrict__ X,
                                               const unsigned short* __restrict__ Wt,
                                               const float* __restrict__ bias,
                                               unsigned short* __restrict__ Y,
                                               float* __restrict__ stats) {
    const int wave = threadIdx.x >> 5;
    const int lane = threadIdx.x & 31;
    const int grp  = lane >> 4;          // K-half select
    const int l16  = lane & 15;
    const size_t rowBase = (size_t)blockIdx.x * 128 + wave * 16;
    const int    colBase = blockIdx.y * 64;

    const unsigned short* xrow = X + (rowBase + l16) * CINP;
    v8f acc[4] = {};

#pragma unroll
    for (int kb = 0; kb < CINP; kb += 32) {
        // A 16x32 bf16 fragment: elems 0-7 = K[kb+grp*16 ..],
        // elems 8-15 = K[kb+grp*16+8 ..]
        v8u lo = *(const v8u*)(xrow + kb + grp * 16);
        v8u hi = *(const v8u*)(xrow + kb + grp * 16 + 8);
        v16u a;
#pragma unroll
        for (int i = 0; i < 8; ++i) { a[i] = lo[i]; a[i + 8] = hi[i]; }
        __builtin_prefetch(xrow + kb + 32, 0, 1);
#pragma unroll
        for (int t = 0; t < 4; ++t) {
            // B 32x16 fragment: N = l16, K = kb + grp*16 + i  (contiguous W row)
            const unsigned short* wrow =
                Wt + (size_t)(colBase + t * 16 + l16) * CINP + kb + grp * 16;
            v16u bfrag = *(const v16u*)wrow;
            acc[t] = __builtin_amdgcn_wmma_f32_16x16x32_bf16(
                false, __builtin_bit_cast(v16bf, a),
                false, __builtin_bit_cast(v16bf, bfrag),
                (short)0, acc[t], false, false);
        }
    }

#pragma unroll
    for (int t = 0; t < 4; ++t) {
        const int n = colBase + t * 16 + l16;
        const float bv = bias[n];
        float s = 0.0f, sq = 0.0f;
#pragma unroll
        for (int r = 0; r < 8; ++r) {
            float v = acc[t][r] + bv;
            size_t rr = rowBase + r + grp * 8;   // C/D layout: lanes16-31 -> M+8
            Y[rr * COUT + n] = f2bf(v);
            s += v; sq += v * v;
        }
        // lanes L and L+16 hold the same channel n: pair-reduce then atomics
        s  += __shfl_xor(s, 16, 32);
        sq += __shfl_xor(sq, 16, 32);
        if (grp == 0) {
            atomicAdd(&stats[n], s);
            atomicAdd(&stats[COUT + n], sq);
        }
    }
}

// scale/shift from batch stats: ss[c] = gamma*rsqrt(var+eps), ss[256+c] = beta - mu*scale
__global__ void bn_finalize(const float* __restrict__ stats,
                            const float* __restrict__ gamma,
                            const float* __restrict__ beta,
                            float* __restrict__ ss, int C, float invP) {
    int c = threadIdx.x;
    if (c >= C) return;
    float mu  = stats[c] * invP;
    float var = stats[C + c] * invP - mu * mu;
    float sc  = gamma[c] * rsqrtf(var + 1e-5f);
    ss[c]       = sc;
    ss[256 + c] = beta[c] - mu * sc;
}

// BN apply + ReLU -> bf16 next-layer input (C == 128 for both inner layers)
__global__ void bn_apply_relu(const unsigned short* __restrict__ Y,
                              const float* __restrict__ ss,
                              unsigned short* __restrict__ Xn) {
    size_t i = (size_t)blockIdx.x * 256 + threadIdx.x;   // < P*128
    int c = (int)(i & 127);
    float v = fmaf(bf2f(Y[i]), ss[c], ss[256 + c]);
    v = v > 0.0f ? v : 0.0f;
    Xn[i] = f2bf(v);
}

// Final: BN + ReLU + max over K, write out (B,256,M)
__global__ __launch_bounds__(256) void bn_relu_maxpool(const unsigned short* __restrict__ Y,
                                                       const float* __restrict__ ss,
                                                       float* __restrict__ out) {
    const int c  = threadIdx.x;       // 0..255
    const int cm = blockIdx.x;        // 0..B*M-1
    const float sc = ss[c], sh = ss[256 + c];
    const unsigned short* yr = Y + ((size_t)cm * Kn) * 256 + c;
    float best = 0.0f;                // relu() >= 0, so 0 is the identity
#pragma unroll 4
    for (int k = 0; k < Kn; ++k) {
        float v = fmaf(bf2f(yr[(size_t)k * 256]), sc, sh);
        v = v > 0.0f ? v : 0.0f;
        best = v > best ? v : best;
    }
    int b = cm >> 10, m = cm & 1023;
    out[((size_t)b * 256 + c) * Mn + m] = best;
}

// ---------------------------------------------------------------------------
extern "C" void kernel_launch(void* const* d_in, const int* in_sizes, int n_in,
                              void* d_out, int out_size, void* d_ws, size_t ws_size,
                              hipStream_t stream) {
    const float* feat   = (const float*)d_in[0];
    const float* coords = (const float*)d_in[1];
    const float* cond   = (const float*)d_in[2];
    const float* w0 = (const float*)d_in[3],  *b0 = (const float*)d_in[4];
    const float* g0 = (const float*)d_in[5],  *be0 = (const float*)d_in[6];
    const float* w1 = (const float*)d_in[7],  *b1 = (const float*)d_in[8];
    const float* g1 = (const float*)d_in[9],  *be1 = (const float*)d_in[10];
    const float* w2 = (const float*)d_in[11], *b2 = (const float*)d_in[12];
    const float* g2 = (const float*)d_in[13], *be2 = (const float*)d_in[14];
    float* out = (float*)d_out;

    // workspace layout (256B aligned)
    char* ws = (char*)d_ws;
    size_t off = 0;
    auto alloc = [&](size_t bytes) {
        size_t r = off;
        off = (off + bytes + 255) & ~(size_t)255;
        return r;
    };
    unsigned short* featT = (unsigned short*)(ws + alloc((size_t)Bn * Nn * CIN * 2));
    float* centersWS      = (float*)(ws + alloc((size_t)Bn * Mn * 3 * 4));
    float* stats          = (float*)(ws + alloc(512 * 4));
    float* ss             = (float*)(ws + alloc(512 * 4));
    unsigned short* W0p   = (unsigned short*)(ws + alloc(128 * 96 * 2));
    unsigned short* W1p   = (unsigned short*)(ws + alloc(128 * 128 * 2));
    unsigned short* W2p   = (unsigned short*)(ws + alloc(256 * 128 * 2));
    unsigned short* XA    = (unsigned short*)(ws + alloc((size_t)Pn * 128 * 2));
    unsigned short* XB    = (unsigned short*)(ws + alloc((size_t)Pn * 128 * 2));
    unsigned short* Ybuf  = (unsigned short*)(ws + alloc((size_t)Pn * 256 * 2));
    (void)ws_size; (void)in_sizes; (void)n_in; (void)out_size;

    const size_t OUT0 = (size_t)Bn * 256 * Mn;   // main output
    const size_t OUT1 = (size_t)Bn * 3 * Mn;     // centers
    const float invP = 1.0f / (float)Pn;

    // stage inputs
    feat_transpose_bf16<<<(Bn * CIN * Nn) / 256, 256, 0, stream>>>(feat, featT);
    pack_weights<<<(128 * 96 + 255) / 256, 256, 0, stream>>>(w0, W0p, 128, 67, 96);
    pack_weights<<<(128 * 128 + 255) / 256, 256, 0, stream>>>(w1, W1p, 128, 128, 128);
    pack_weights<<<(256 * 128 + 255) / 256, 256, 0, stream>>>(w2, W2p, 256, 128, 128);

    fps_kernel<<<Bn, 1024, 0, stream>>>(coords, centersWS, out + OUT0);
    ball_pack<<<(Bn * Mn) / 8, 256, 0, stream>>>(coords, centersWS, featT, XA);
    copy_f32<<<2, 256, 0, stream>>>(cond, out + OUT0 + OUT1, Bn * CIN);

    // layer 0: 96 -> 128   (XA -> Ybuf -> XB)
    zero_f32<<<2, 256, 0, stream>>>(stats, 512);
    gemm_bn<96, 128><<<dim3(Pn / 128, 2), 256, 0, stream>>>(XA, W0p, b0, Ybuf, stats);
    bn_finalize<<<1, 128, 0, stream>>>(stats, g0, be0, ss, 128, invP);
    bn_apply_relu<<<(size_t)Pn * 128 / 256, 256, 0, stream>>>(Ybuf, ss, XB);

    // layer 1: 128 -> 128  (XB -> Ybuf -> XA)
    zero_f32<<<2, 256, 0, stream>>>(stats, 512);
    gemm_bn<128, 128><<<dim3(Pn / 128, 2), 256, 0, stream>>>(XB, W1p, b1, Ybuf, stats);
    bn_finalize<<<1, 128, 0, stream>>>(stats, g1, be1, ss, 128, invP);
    bn_apply_relu<<<(size_t)Pn * 128 / 256, 256, 0, stream>>>(Ybuf, ss, XA);

    // layer 2: 128 -> 256  (XA -> Ybuf -> out with maxpool)
    zero_f32<<<2, 256, 0, stream>>>(stats, 512);
    gemm_bn<128, 256><<<dim3(Pn / 128, 4), 256, 0, stream>>>(XA, W2p, b2, Ybuf, stats);
    bn_finalize<<<1, 256, 0, stream>>>(stats, g2, be2, ss, 256, invP);
    bn_relu_maxpool<<<Bn * Mn, 256, 0, stream>>>(Ybuf, ss, out);
}
